// WindowAttention_47656957116638
// MI455X (gfx1250) — matrix operand order repaired and checked by reading, not compiled, not run on hardware
//
#include <hip/hip_runtime.h>
#include <hip/hip_bf16.h>
#include <stdint.h>

typedef __bf16 bf16;
typedef __attribute__((ext_vector_type(16))) __bf16 bf16x16;
typedef __attribute__((ext_vector_type(8)))  __bf16 bf16x8;
typedef __attribute__((ext_vector_type(4)))  __bf16 bf16x4;
typedef __attribute__((ext_vector_type(8)))  float  f32x8;
typedef __attribute__((ext_vector_type(4)))  float  f32x4;
typedef __attribute__((ext_vector_type(4)))  int    i32x4;

#define DIM_   768
#define HEADS_ 12
#define NTOK_  64
#define BWIN_  1024
#define MTOT_  (BWIN_ * NTOK_)   // 65536 rows
#define APITCH 40                // 32 + 8 bf16 pad (80B rows, 16B aligned)

// Async global->LDS (CDNA5 GLOBAL_LOAD_ASYNC_TO_LDS_B128, ASYNCcnt-tracked).
#if defined(__AMDGCN__) && defined(__gfx1250__) && \
    __has_builtin(__builtin_amdgcn_global_load_async_to_lds_b128) && \
    __has_builtin(__builtin_amdgcn_s_wait_asynccnt)
#define ASYNC_LDS 1
#else
#define ASYNC_LDS 0
#endif

// ---------- scalar helpers ----------
__device__ __forceinline__ bf16 f2bf(float f) {
  union { float f; unsigned u; } a; a.f = f;
  unsigned r = a.u + 0x7FFFu + ((a.u >> 16) & 1u);   // round-to-nearest-even
  union { unsigned short s; bf16 b; } o; o.s = (unsigned short)(r >> 16);
  return o.b;
}
__device__ __forceinline__ float bf2f(bf16 b) {
  union { unsigned short s; bf16 b; } i; i.b = b;
  union { unsigned u; float f; } o; o.u = ((unsigned)i.s) << 16;
  return o.f;
}
__device__ __forceinline__ f32x8 zero8() {
  f32x8 z;
#pragma unroll
  for (int i = 0; i < 8; ++i) z[i] = 0.f;
  return z;
}

// 16-byte tile copy: async DMA into LDS when available, else sync via VGPRs.
__device__ __forceinline__ void cp16(bf16* l, const bf16* g) {
#if ASYNC_LDS
  __builtin_amdgcn_global_load_async_to_lds_b128(
      (__attribute__((address_space(1))) i32x4*)(bf16*)g,
      (__attribute__((address_space(3))) i32x4*)l, 0, 0);
#else
  *(bf16x8*)l = *(const bf16x8*)g;
#endif
}
__device__ __forceinline__ void wait_async_le(int n) {
#if ASYNC_LDS
  if (n == 0)      __builtin_amdgcn_s_wait_asynccnt(0);
  else if (n == 4) __builtin_amdgcn_s_wait_asynccnt(4);
  else             __builtin_amdgcn_s_wait_asynccnt(8);
#else
  (void)n;
#endif
}

// ---------- WMMA fragment loads (layouts per CDNA5 ISA 7.12.2, wave32) ----------
// A 16x32 bf16: lanes 0-15 row M=lane hold K {k..k+7, k+16..k+23};
//               lanes 16-31 row M=lane-16 hold K {k+8..k+15, k+24..k+31}.
__device__ __forceinline__ bf16x16 ldsA(const bf16* base, int pitch, int m, int k, int lane) {
  const bf16* p = base + (size_t)(m + (lane & 15)) * pitch + k + ((lane >> 4) ? 8 : 0);
  bf16x8 lo = *(const bf16x8*)p;
  bf16x8 hi = *(const bf16x8*)(p + 16);
  bf16x16 f;
#pragma unroll
  for (int i = 0; i < 8; ++i) { f[i] = lo[i]; f[i + 8] = hi[i]; }
  return f;
}
// B 32x16 bf16 from transposed (N-major) tile: lane n<16 col n holds K {k..k+7, k+16..k+23},
// lanes 16-31 col n-16 hold K {k+8..k+15, k+24..k+31} (matches A).
__device__ __forceinline__ bf16x16 ldsB(const bf16* baseT, int pitch, int n, int k, int lane) {
  const bf16* p = baseT + (size_t)(n + (lane & 15)) * pitch + k + ((lane >> 4) ? 8 : 0);
  bf16x8 lo = *(const bf16x8*)p;
  bf16x8 hi = *(const bf16x8*)(p + 16);
  bf16x16 f;
#pragma unroll
  for (int i = 0; i < 8; ++i) { f[i] = lo[i]; f[i + 8] = hi[i]; }
  return f;
}
__device__ __forceinline__ f32x8 wmma_bf16(bf16x16 a, bf16x16 b, f32x8 c) {
  return __builtin_amdgcn_wmma_f32_16x16x32_bf16(false, a, false, b, (short)0, c, false, false);
}

// ---------- prep kernels ----------
__global__ __launch_bounds__(256) void cvt_x(const float* __restrict__ x,
                                             bf16* __restrict__ xb, int n4) {
  int i = blockIdx.x * 256 + threadIdx.x;
  if (i >= n4) return;
  f32x4 v = *(const f32x4*)&x[(size_t)i * 4];
  bf16x4 o;
#pragma unroll
  for (int e = 0; e < 4; ++e) o[e] = f2bf(v[e]);
  *(bf16x4*)&xb[(size_t)i * 4] = o;
}

// wt[n][k] = bf16(w[k][n]), 768x768
__global__ __launch_bounds__(256) void transpose_w(const float* __restrict__ w,
                                                   bf16* __restrict__ wt) {
  __shared__ float tile[32][33];
  int k0 = blockIdx.x * 32, n0 = blockIdx.y * 32;
  int tx = threadIdx.x & 31, ty = threadIdx.x >> 5;   // 32 x 8
#pragma unroll
  for (int i = 0; i < 32; i += 8)
    tile[ty + i][tx] = w[(size_t)(k0 + ty + i) * DIM_ + n0 + tx];
  __syncthreads();
#pragma unroll
  for (int i = 0; i < 32; i += 8)
    wt[(size_t)(n0 + ty + i) * DIM_ + k0 + tx] = f2bf(tile[tx][ty + i]);
}

// CPB MLP: tbl[i][h] = (relu(coords[i] @ w1 + b1) @ w2 + b2)[h], i < 343
__global__ __launch_bounds__(256) void cpb_mlp(const float* __restrict__ coords,
                                               const float* __restrict__ w1,
                                               const float* __restrict__ b1,
                                               const float* __restrict__ w2,
                                               const float* __restrict__ b2,
                                               float* __restrict__ tbl) {
  int i = blockIdx.x * 256 + threadIdx.x;
  if (i >= 343) return;
  float c0 = coords[i * 3 + 0], c1 = coords[i * 3 + 1], c2 = coords[i * 3 + 2];
  float acc[HEADS_];
#pragma unroll
  for (int h = 0; h < HEADS_; ++h) acc[h] = b2[h];
  for (int j = 0; j < 512; ++j) {
    float s = fmaxf(c0 * w1[j] + c1 * w1[512 + j] + c2 * w1[1024 + j] + b1[j], 0.f);
#pragma unroll
    for (int h = 0; h < HEADS_; ++h) acc[h] += s * w2[j * HEADS_ + h];
  }
#pragma unroll
  for (int h = 0; h < HEADS_; ++h) tbl[i * HEADS_ + h] = acc[h];
}

// bias16[h][n][m] = 16*sigmoid(tbl[rpi[n*64+m]][h])
__global__ __launch_bounds__(256) void cpb_bias(const float* __restrict__ tbl,
                                                const int* __restrict__ rpi,
                                                float* __restrict__ bias16) {
  int i = blockIdx.x * 256 + threadIdx.x;
  if (i >= HEADS_ * NTOK_ * NTOK_) return;
  int h = i / (NTOK_ * NTOK_), nm = i % (NTOK_ * NTOK_);
  float v = tbl[rpi[nm] * HEADS_ + h];
  bias16[i] = 16.f / (1.f + __expf(-v));
}

// ---------- fused QKV GEMM: 128x128 tile, 8 waves, 3 weights share A,
// ---------- double-buffered async global->LDS staging ----------
__global__ __launch_bounds__(256) void qkv_gemm(
    const bf16* __restrict__ xb,                                  // (65536, 768)
    const bf16* __restrict__ wtq, const bf16* __restrict__ wtk,   // (768 n, 768 k)
    const bf16* __restrict__ wtv,
    const float* __restrict__ qbias, const float* __restrict__ vbias,
    bf16* __restrict__ qo, bf16* __restrict__ ko, bf16* __restrict__ vo) {
  __shared__ bf16 As[2][128][APITCH];
  __shared__ bf16 Bs[2][3][128][APITCH];
  const int tid = threadIdx.x, lane = tid & 31, wave = tid >> 5;
  const int m0 = blockIdx.x * 128, n0 = blockIdx.y * 128;
  const int wm = (wave & 3) * 32, wn = (wave >> 2) * 64;
  const int r = (tid * 2) >> 2, c0 = ((tid * 2) & 3) * 8;   // staging coords

  f32x8 acc[3][2][4];
#pragma unroll
  for (int w3 = 0; w3 < 3; ++w3)
#pragma unroll
    for (int mi = 0; mi < 2; ++mi)
#pragma unroll
      for (int j = 0; j < 4; ++j) acc[w3][mi][j] = zero8();

  auto stage = [&](int bufi, int kt) {               // 8 x b128 per thread
#pragma unroll
    for (int i = 0; i < 2; ++i) {
      int rr = r + ((c0 + i * 8) >> 5);              // i advances 8 cols; wraps row
      int cc = (c0 + i * 8) & 31;
      cp16(&As[bufi][rr][cc],    &xb [(size_t)(m0 + rr) * DIM_ + kt + cc]);
      cp16(&Bs[bufi][0][rr][cc], &wtq[(size_t)(n0 + rr) * DIM_ + kt + cc]);
      cp16(&Bs[bufi][1][rr][cc], &wtk[(size_t)(n0 + rr) * DIM_ + kt + cc]);
      cp16(&Bs[bufi][2][rr][cc], &wtv[(size_t)(n0 + rr) * DIM_ + kt + cc]);
    }
  };
  auto compute = [&](int bufi) {
    bf16x16 a0 = ldsA(&As[bufi][0][0], APITCH, wm, 0, lane);
    bf16x16 a1 = ldsA(&As[bufi][0][0], APITCH, wm + 16, 0, lane);
#pragma unroll
    for (int w3 = 0; w3 < 3; ++w3) {
#pragma unroll
      for (int j = 0; j < 4; ++j) {
        bf16x16 b = ldsB(&Bs[bufi][w3][0][0], APITCH, wn + j * 16, 0, lane);
        acc[w3][0][j] = wmma_bf16(a0, b, acc[w3][0][j]);
        acc[w3][1][j] = wmma_bf16(a1, b, acc[w3][1][j]);
      }
    }
  };

  stage(0, 0);
  int buf = 0;
  for (int kt = 32; kt < DIM_; kt += 32) {
    stage(buf ^ 1, kt);        // prefetch next tile into the other buffer
    wait_async_le(8);          // previous tile's copies have landed in LDS
    __syncthreads();
    compute(buf);
    __syncthreads();           // everyone done reading before overwrite
    buf ^= 1;
  }
  wait_async_le(0);
  __syncthreads();
  compute(buf);

  // epilogue: scatter bf16 into (b, h, tok, dh); q += q_b, v += v_b, k no bias
#pragma unroll
  for (int w3 = 0; w3 < 3; ++w3) {
    bf16* outp = (w3 == 0) ? qo : (w3 == 1) ? ko : vo;
#pragma unroll
    for (int mi = 0; mi < 2; ++mi)
#pragma unroll
      for (int j = 0; j < 4; ++j) {
        int Nc = n0 + wn + j * 16 + (lane & 15);
        int h = Nc >> 6, dh = Nc & 63;
        float bv = (w3 == 0) ? qbias[Nc] : (w3 == 2) ? vbias[Nc] : 0.f;
#pragma unroll
        for (int rr = 0; rr < 8; ++rr) {
          int Mr = m0 + wm + mi * 16 + rr + ((lane >> 4) ? 8 : 0);
          int bw = Mr >> 6, tok = Mr & 63;
          size_t off = (((size_t)bw * HEADS_ + h) * NTOK_ + tok) * 64 + dh;
          outp[off] = f2bf(acc[w3][mi][j][rr] + bv);
        }
      }
  }
}

// ---------- attention: one workgroup per (window, head) ----------
__global__ __launch_bounds__(256) void attn_kernel(
    const bf16* __restrict__ qbuf, const bf16* __restrict__ kbuf,
    const bf16* __restrict__ vbuf, const float* __restrict__ logit_scale,
    const float* __restrict__ bias16, bf16* __restrict__ pbuf) {
  __shared__ bf16 Qs[64][72];
  __shared__ bf16 Ks[64][72];
  __shared__ bf16 Vts[64][72];   // V transposed: [dh][tok]
  __shared__ bf16 Ps[64][72];
  __shared__ float Sf[64][68];
  __shared__ float qn[64], kn[64];

  const int bh = blockIdx.x;
  const int h = bh % HEADS_, bw = bh / HEADS_;
  const size_t base = (size_t)bh * 64 * 64;
  const int tid = threadIdx.x, lane = tid & 31, wave = tid >> 5;

#pragma unroll
  for (int i = 0; i < 2; ++i) {                      // stage Q, K, V^T
    int idx = tid * 2 + i;
    int r = idx >> 3, c = (idx & 7) * 8;
    *(bf16x8*)&Qs[r][c] = *(const bf16x8*)&qbuf[base + r * 64 + c];
    *(bf16x8*)&Ks[r][c] = *(const bf16x8*)&kbuf[base + r * 64 + c];
    bf16x8 v = *(const bf16x8*)&vbuf[base + r * 64 + c];
#pragma unroll
    for (int e = 0; e < 8; ++e) Vts[c + e][r] = v[e];
  }
  __syncthreads();
  if (tid < 128) {                                   // row norms of Q, K
    int r = tid & 63;
    const bf16* row = (tid < 64) ? &Qs[r][0] : &Ks[r][0];
    float s = 0.f;
#pragma unroll
    for (int c = 0; c < 64; ++c) { float x = bf2f(row[c]); s += x * x; }
    if (tid < 64) qn[r] = sqrtf(s); else kn[r] = sqrtf(s);
  }
  __syncthreads();
  const float ls = fmaxf(logit_scale[h], 0.01f);

#pragma unroll
  for (int t = 0; t < 2; ++t) {                      // S = Q K^T (16 tiles / 8 waves)
    int tile = wave * 2 + t;
    int ti = (tile >> 2) * 16, tj = (tile & 3) * 16;
    f32x8 acc = zero8();
    acc = wmma_bf16(ldsA(&Qs[0][0], 72, ti, 0, lane),  ldsB(&Ks[0][0], 72, tj, 0, lane),  acc);
    acc = wmma_bf16(ldsA(&Qs[0][0], 72, ti, 32, lane), ldsB(&Ks[0][0], 72, tj, 32, lane), acc);
    int Nc = tj + (lane & 15);
    float knv = kn[Nc];
#pragma unroll
    for (int r = 0; r < 8; ++r) {
      int Mr = ti + r + ((lane >> 4) ? 8 : 0);
      float denom = fmaxf(qn[Mr] * knv, 1e-6f);
      Sf[Mr][Nc] = acc[r] * ls / denom + bias16[(h * 64 + Mr) * 64 + Nc];
    }
  }
  __syncthreads();
  if (tid < 64) {                                    // softmax per row
    int r = tid;
    float m = -1e30f;
#pragma unroll
    for (int c = 0; c < 64; ++c) m = fmaxf(m, Sf[r][c]);
    float s = 0.f;
#pragma unroll
    for (int c = 0; c < 64; ++c) { float e = __expf(Sf[r][c] - m); Sf[r][c] = e; s += e; }
    float inv = 1.f / s;
#pragma unroll
    for (int c = 0; c < 64; ++c) Ps[r][c] = f2bf(Sf[r][c] * inv);
  }
  __syncthreads();
#pragma unroll
  for (int t = 0; t < 2; ++t) {                      // O = P V
    int tile = wave * 2 + t;
    int ti = (tile >> 2) * 16, tj = (tile & 3) * 16;
    f32x8 acc = zero8();
    acc = wmma_bf16(ldsA(&Ps[0][0], 72, ti, 0, lane),  ldsB(&Vts[0][0], 72, tj, 0, lane),  acc);
    acc = wmma_bf16(ldsA(&Ps[0][0], 72, ti, 32, lane), ldsB(&Vts[0][0], 72, tj, 32, lane), acc);
    int dh = tj + (lane & 15);
#pragma unroll
    for (int r = 0; r < 8; ++r) {
      int tok = ti + r + ((lane >> 4) ? 8 : 0);
      pbuf[((size_t)bw * NTOK_ + tok) * DIM_ + h * 64 + dh] = f2bf(acc[r]);
    }
  }
}

// ---------- output projection GEMM, double-buffered async staging ----------
__global__ __launch_bounds__(256) void proj_gemm(
    const bf16* __restrict__ ab, const bf16* __restrict__ wtp,
    const float* __restrict__ bias, float* __restrict__ out) {
  __shared__ bf16 As[2][128][APITCH];
  __shared__ bf16 Bs[2][128][APITCH];
  const int tid = threadIdx.x, lane = tid & 31, wave = tid >> 5;
  const int m0 = blockIdx.x * 128, n0 = blockIdx.y * 128;
  const int wm = (wave & 3) * 32, wn = (wave >> 2) * 64;
  const int r = (tid * 2) >> 2, c0 = ((tid * 2) & 3) * 8;

  f32x8 acc[2][4];
#pragma unroll
  for (int mi = 0; mi < 2; ++mi)
#pragma unroll
    for (int j = 0; j < 4; ++j) acc[mi][j] = zero8();

  auto stage = [&](int bufi, int kt) {               // 4 x b128 per thread
#pragma unroll
    for (int i = 0; i < 2; ++i) {
      int rr = r + ((c0 + i * 8) >> 5);
      int cc = (c0 + i * 8) & 31;
      cp16(&As[bufi][rr][cc], &ab [(size_t)(m0 + rr) * DIM_ + kt + cc]);
      cp16(&Bs[bufi][rr][cc], &wtp[(size_t)(n0 + rr) * DIM_ + kt + cc]);
    }
  };
  auto compute = [&](int bufi) {
    bf16x16 a0 = ldsA(&As[bufi][0][0], APITCH, wm, 0, lane);
    bf16x16 a1 = ldsA(&As[bufi][0][0], APITCH, wm + 16, 0, lane);
#pragma unroll
    for (int j = 0; j < 4; ++j) {
      bf16x16 b = ldsB(&Bs[bufi][0][0], APITCH, wn + j * 16, 0, lane);
      acc[0][j] = wmma_bf16(a0, b, acc[0][j]);
      acc[1][j] = wmma_bf16(a1, b, acc[1][j]);
    }
  };

  stage(0, 0);
  int buf = 0;
  for (int kt = 32; kt < DIM_; kt += 32) {
    stage(buf ^ 1, kt);
    wait_async_le(4);
    __syncthreads();
    compute(buf);
    __syncthreads();
    buf ^= 1;
  }
  wait_async_le(0);
  __syncthreads();
  compute(buf);

#pragma unroll
  for (int mi = 0; mi < 2; ++mi)
#pragma unroll
    for (int j = 0; j < 4; ++j) {
      int Nc = n0 + wn + j * 16 + (lane & 15);
      float bv = bias[Nc];
#pragma unroll
      for (int rr = 0; rr < 8; ++rr) {
        int Mr = m0 + wm + mi * 16 + rr + ((lane >> 4) ? 8 : 0);
        out[(size_t)Mr * DIM_ + Nc] = acc[mi][j][rr] + bv;
      }
    }
}

// ---------- launch ----------
extern "C" void kernel_launch(void* const* d_in, const int* in_sizes, int n_in,
                              void* d_out, int out_size, void* d_ws, size_t ws_size,
                              hipStream_t stream) {
  (void)in_sizes; (void)n_in; (void)out_size; (void)ws_size;
  const float* x       = (const float*)d_in[0];
  const float* q_w     = (const float*)d_in[1];
  const float* q_b     = (const float*)d_in[2];
  const float* k_w     = (const float*)d_in[3];
  const float* v_w     = (const float*)d_in[4];
  const float* v_b     = (const float*)d_in[5];
  const float* cpb_w1  = (const float*)d_in[6];
  const float* cpb_b1  = (const float*)d_in[7];
  const float* cpb_w2  = (const float*)d_in[8];
  const float* cpb_b2  = (const float*)d_in[9];
  const float* lscale  = (const float*)d_in[10];
  const float* proj_w  = (const float*)d_in[11];
  const float* proj_b  = (const float*)d_in[12];
  const float* coords  = (const float*)d_in[13];
  const int*   rpi     = (const int*)d_in[14];
  float* out = (float*)d_out;

  const size_t NTOT = (size_t)MTOT_ * DIM_;          // 50,331,648
  const size_t WB   = (size_t)DIM_ * DIM_ * 2;       // bf16 weight bytes

  char* ws = (char*)d_ws;
  auto take = [&](size_t bytes) { char* p = ws; ws += (bytes + 255) & ~(size_t)255; return p; };
  bf16* xb    = (bf16*)take(NTOT * 2);
  bf16* wtq   = (bf16*)take(WB);
  bf16* wtk   = (bf16*)take(WB);
  bf16* wtv   = (bf16*)take(WB);
  bf16* wtp   = (bf16*)take(WB);
  bf16* qbuf  = (bf16*)take(NTOT * 2);
  bf16* kbuf  = (bf16*)take(NTOT * 2);
  bf16* vbuf  = (bf16*)take(NTOT * 2);
  bf16* pbuf  = (bf16*)take(NTOT * 2);
  float* tbl  = (float*)take(343 * HEADS_ * 4);
  float* b16  = (float*)take(HEADS_ * NTOK_ * NTOK_ * 4);

  cvt_x<<<(int)(NTOT / 4 / 256), 256, 0, stream>>>(x, xb, (int)(NTOT / 4));
  dim3 tg(DIM_ / 32, DIM_ / 32);
  transpose_w<<<tg, 256, 0, stream>>>(q_w,    wtq);
  transpose_w<<<tg, 256, 0, stream>>>(k_w,    wtk);
  transpose_w<<<tg, 256, 0, stream>>>(v_w,    wtv);
  transpose_w<<<tg, 256, 0, stream>>>(proj_w, wtp);
  cpb_mlp<<<2, 256, 0, stream>>>(coords, cpb_w1, cpb_b1, cpb_w2, cpb_b2, tbl);
  cpb_bias<<<(HEADS_ * NTOK_ * NTOK_ + 255) / 256, 256, 0, stream>>>(tbl, rpi, b16);

  qkv_gemm<<<dim3(MTOT_ / 128, DIM_ / 128), 256, 0, stream>>>(
      xb, wtq, wtk, wtv, q_b, v_b, qbuf, kbuf, vbuf);
  attn_kernel<<<BWIN_ * HEADS_, 256, 0, stream>>>(qbuf, kbuf, vbuf, lscale, b16, pbuf);
  proj_gemm<<<dim3(MTOT_ / 128, DIM_ / 128), 256, 0, stream>>>(pbuf, wtp, proj_b, out);
}
